// Decoder_80857054314781
// MI455X (gfx1250) — matrix operand order repaired and checked by reading, not compiled, not run on hardware
//
#include <hip/hip_runtime.h>
#include <hip/hip_bf16.h>
#include <math.h>

// ---------------- problem constants ----------------
#define B_   128
#define S_   64
#define H_   1024
#define E_   512
#define V_   32000
#define T_   31
#define H4_  256
#define ZW_  2560   // [e(512) | ctx(1024) | h(1024)]
#define G4H_ 4096   // 4*H

#define NSUB 8      // 8 x 16 = 128-wide N tile per block
#define KSTEP 32

typedef __bf16 bf16;
typedef __attribute__((ext_vector_type(8)))  __bf16 v8bf;
typedef __attribute__((ext_vector_type(16))) __bf16 v16bf;
typedef __attribute__((ext_vector_type(8)))  float  v8f;

union ABFrag { v8bf h[2]; v16bf w; };

__device__ __forceinline__ bf16 f2bf(float f) {
    union { float f; unsigned u; } v; v.f = f;
    unsigned r = v.u + 0x7FFFu + ((v.u >> 16) & 1u);   // round-to-nearest-even
    unsigned short hs = (unsigned short)(r >> 16);
    bf16 out; __builtin_memcpy(&out, &hs, 2);
    return out;
}

__device__ __forceinline__ float sigmoidf_(float x) {
    return 1.0f / (1.0f + __expf(-x));
}

// ---- CDNA5 async global->LDS copy (16B per lane), ASYNCcnt-tracked ----
// VDST = 32-bit LDS byte address (low 32 bits of the generic pointer of a
// __shared__ object ARE the LDS offset per the aperture mapping).
template <int OFF>
__device__ __forceinline__ void async_ldsb128(const bf16* g, bf16* l) {
    unsigned loff = (unsigned)(unsigned long long)(void*)l;
    asm volatile("global_load_async_to_lds_b128 %0, %1, off offset:%2"
                 :: "v"(loff), "v"((unsigned long long)(const void*)g), "i"(OFF)
                 : "memory");
}

template <int N>
__device__ __forceinline__ void wait_asynccnt_le() {
    asm volatile("s_wait_asynccnt %0" :: "i"(N) : "memory");
}

// =====================================================================
// WMMA GEMM:  C[M,N] = A[M,K](bf16,row-major,lda) * W[N,K]^T
//                      (bf16,row-major,ldw) + bias[N]
// 256 threads = 8 waves. Block computes a 128(M) x 128(N) tile.
// The 32K x 128N B tile (8 KB) is async-copied global->LDS, double
// buffered; each wave owns one 16-row M tile and software-pipelines
// its A fragment.
// =====================================================================
__global__ __launch_bounds__(256) void wmma_gemm_kernel(
    const bf16* __restrict__ A, int lda,
    const bf16* __restrict__ W, int ldw,
    const float* __restrict__ bias,
    float* __restrict__ C, long long ldc,
    int M, int N, int K)
{
    // [2 buffers][NSUB subtiles][32 lanes][16 bf16] = 16 KB
    __shared__ __attribute__((aligned(32))) bf16 lds[2][NSUB * 32 * 16];

    const int tid  = threadIdx.x;
    const int lane = tid & 31;
    const int wv   = tid >> 5;

    const int numMTiles = M >> 4;
    const int mGroups   = (numMTiles + 7) >> 3;
    const int nTileIdx  = blockIdx.x / mGroups;
    const int mGroup    = blockIdx.x % mGroups;
    const int nBase     = nTileIdx * (NSUB * 16);
    const int mTile     = mGroup * 8 + wv;
    const bool active   = (mTile < numMTiles);      // wave-uniform

    v8f acc[NSUB] = {};

    const int mRow  = (mTile << 4) + (lane & 15);
    const int koffA = (lane >> 4) << 4;             // 0 or 16
    const bf16* aBase = A + (long long)mRow * lda;

    // staging map: each of 256 threads owns one 32B B-fragment slot
    const int sj    = tid >> 5;                     // n sub-tile 0..7
    const int sn    = nBase + (sj << 4) + (lane & 15);
    const int skoff = (lane >> 4) << 4;             // 0 or 16 (K offset)
    const bf16* wSrc0 = W + (long long)sn * ldw + skoff;
    bf16* const ldsDst0 = &lds[0][(sj * 32 + lane) * 16];
    bf16* const ldsDst1 = &lds[1][(sj * 32 + lane) * 16];

    const int nk = K / KSTEP;

    // prologue: async-stage tile 0, preload A fragment 0
    async_ldsb128<0>(wSrc0, ldsDst0);
    async_ldsb128<16>(wSrc0, ldsDst0);
    ABFrag aCur, aNxt;
    if (active) {
        aCur.h[0] = *(const v8bf*)(aBase + koffA);
        aCur.h[1] = *(const v8bf*)(aBase + 8 + koffA);
    }

    int cur = 0;
    for (int ki = 0; ki < nk; ++ki) {
        __syncthreads();                         // buf[cur^1] free to overwrite
        if (ki + 1 < nk) {                       // stage next tile (<=1 tile in flight)
            const bf16* ws = wSrc0 + (long long)(ki + 1) * KSTEP;
            bf16* ld = (cur == 0) ? ldsDst1 : ldsDst0;
            async_ldsb128<0>(ws, ld);
            async_ldsb128<16>(ws, ld);
            wait_asynccnt_le<2>();               // tile ki has landed (this wave)
        } else {
            wait_asynccnt_le<0>();
        }
        __syncthreads();                         // tile ki visible to all waves

        if (active) {
            // software-pipelined A fragment for next K chunk (clamped)
            const int knext = (ki + 1 < nk) ? (ki + 1) * KSTEP : ki * KSTEP;
            aNxt.h[0] = *(const v8bf*)(aBase + knext + koffA);
            aNxt.h[1] = *(const v8bf*)(aBase + knext + 8 + koffA);
#pragma unroll
            for (int j = 0; j < NSUB; ++j) {
                v16bf bfr = *(const v16bf*)&lds[cur][(j * 32 + lane) * 16];
                acc[j] = __builtin_amdgcn_wmma_f32_16x16x32_bf16(
                    false, aCur.w, false, bfr, (short)0, acc[j], false, false);
            }
            aCur = aNxt;
        }
        cur ^= 1;
    }

    if (active) {
        const int rBase = (mTile << 4) + ((lane >> 4) << 3);
#pragma unroll
        for (int j = 0; j < NSUB; ++j) {
            const int col = nBase + (j << 4) + (lane & 15);
            const float bb = bias ? bias[col] : 0.0f;
#pragma unroll
            for (int r = 0; r < 8; ++r)
                C[(long long)(rBase + r) * ldc + col] = acc[j][r] + bb;
        }
    }
}

// =====================================================================
// One-time prep kernels
// =====================================================================
__global__ __launch_bounds__(256) void cvt_f32_bf16_kernel(
    const float* __restrict__ src, bf16* __restrict__ dst, long long n)
{
    long long i = (long long)blockIdx.x * blockDim.x + threadIdx.x;
    const long long stride = (long long)gridDim.x * blockDim.x;
    for (; i < n; i += stride) dst[i] = f2bf(src[i]);
}

__global__ __launch_bounds__(256) void pack_wcat_kernel(
    const float* __restrict__ Wih, const float* __restrict__ Whh,
    bf16* __restrict__ Wcat)
{
    const long long total = (long long)G4H_ * ZW_;
    long long i = (long long)blockIdx.x * blockDim.x + threadIdx.x;
    const long long stride = (long long)gridDim.x * blockDim.x;
    for (; i < total; i += stride) {
        const int n = (int)(i / ZW_);
        const int c = (int)(i % ZW_);
        const float v = (c < E_ + H_) ? Wih[(long long)n * (E_ + H_) + c]
                                      : Whh[(long long)n * H_ + (c - (E_ + H_))];
        Wcat[i] = f2bf(v);
    }
}

__global__ __launch_bounds__(256) void bias_sum_kernel(
    const float* __restrict__ bih, const float* __restrict__ bhh,
    float* __restrict__ bg)
{
    const int i = blockIdx.x * blockDim.x + threadIdx.x;
    if (i < G4H_) bg[i] = bih[i] + bhh[i];
}

__global__ __launch_bounds__(256) void init_state_kernel(
    const float* __restrict__ enc_c, const float* __restrict__ enc_h,
    float* __restrict__ c, bf16* __restrict__ z)
{
    const int idx = blockIdx.x * 256 + threadIdx.x;   // B*H = 131072
    const int b = idx >> 10, h = idx & 1023;
    c[idx] = enc_c[idx];
    z[(long long)b * ZW_ + (E_ + H_) + h] = f2bf(enc_h[idx]);
}

// =====================================================================
// Per-step kernels
// =====================================================================
__global__ __launch_bounds__(256) void embed_kernel(
    const int* __restrict__ x_tgt, int t, const float* __restrict__ emb,
    bf16* __restrict__ z)
{
    const int b = blockIdx.x;                  // 128 blocks
    const int tok = x_tgt[b * 32 + t];
    const float* er = emb + (long long)tok * E_;
    bf16* zr = z + (long long)b * ZW_;
    for (int i = threadIdx.x; i < E_; i += blockDim.x) zr[i] = f2bf(er[i]);
}

// per (b,s): softmax over H4 features of (A0[b,s,:] + ae[b,:]), tanh, dot wv
__global__ __launch_bounds__(256) void score_kernel(
    const float* __restrict__ A0, const float* __restrict__ ae,
    const float* __restrict__ wv_w, const float* __restrict__ bv,
    float* __restrict__ score)
{
    const int gw   = blockIdx.x * 8 + (threadIdx.x >> 5);   // 0..B*S-1
    const int lane = threadIdx.x & 31;
    const int b    = gw >> 6;
    const float* row = A0 + (long long)gw * H4_;
    const float* aeb = ae + (long long)b * H4_;

    float v[8];
    float mx = -INFINITY;
#pragma unroll
    for (int i = 0; i < 8; ++i) {
        v[i] = row[lane * 8 + i] + aeb[lane * 8 + i];
        mx = fmaxf(mx, v[i]);
    }
#pragma unroll
    for (int off = 16; off; off >>= 1) mx = fmaxf(mx, __shfl_xor(mx, off, 32));
    float sum = 0.0f;
#pragma unroll
    for (int i = 0; i < 8; ++i) { v[i] = __expf(v[i] - mx); sum += v[i]; }
#pragma unroll
    for (int off = 16; off; off >>= 1) sum += __shfl_xor(sum, off, 32);
    const float inv = 1.0f / sum;
    float acc = 0.0f;
#pragma unroll
    for (int i = 0; i < 8; ++i) acc += tanhf(v[i] * inv) * wv_w[lane * 8 + i];
#pragma unroll
    for (int off = 16; off; off >>= 1) acc += __shfl_xor(acc, off, 32);
    if (lane == 0) score[gw] = acc + bv[0];
}

// ctx[b,h] = sum_s score[b,s] * enc_o[b,s,h] -> z[b, 512+h] (bf16)
__global__ __launch_bounds__(256) void ctx_kernel(
    const float* __restrict__ score, const float* __restrict__ enc_o,
    bf16* __restrict__ z)
{
    const int b = blockIdx.x >> 2;                       // 512 blocks
    const int h = ((blockIdx.x & 3) << 8) + threadIdx.x;
    const float* sc = score + b * S_;
    const float* eo = enc_o + ((long long)b * S_) * H_ + h;
    float acc = 0.0f;
#pragma unroll 8
    for (int s = 0; s < S_; ++s) acc += sc[s] * eo[(long long)s * H_];
    z[(long long)b * ZW_ + E_ + h] = f2bf(acc);
}

__global__ __launch_bounds__(256) void lstm_kernel(
    const float* __restrict__ gates, float* __restrict__ c,
    bf16* __restrict__ z)
{
    const int idx = blockIdx.x * 256 + threadIdx.x;      // B*H
    const int b = idx >> 10, h = idx & 1023;
    const float* g = gates + (long long)b * G4H_;
    const float ig = sigmoidf_(g[h]);
    const float fg = sigmoidf_(g[H_ + h]);
    const float gg = tanhf(g[2 * H_ + h]);
    const float og = sigmoidf_(g[3 * H_ + h]);
    const float cn = fg * c[idx] + ig * gg;
    c[idx] = cn;
    z[(long long)b * ZW_ + (E_ + H_) + h] = f2bf(og * tanhf(cn));
}

// =====================================================================
// Host side
// =====================================================================
static inline void launch_gemm(const bf16* A, int lda, const bf16* W, int ldw,
                               const float* bias, float* C, long long ldc,
                               int M, int N, int K, hipStream_t s)
{
    const int numMTiles = M / 16;
    const int mGroups = (numMTiles + 7) / 8;
    const int nTiles = N / (NSUB * 16);
    hipLaunchKernelGGL(wmma_gemm_kernel, dim3(mGroups * nTiles), dim3(256), 0, s,
                       A, lda, W, ldw, bias, C, ldc, M, N, K);
}

extern "C" void kernel_launch(void* const* d_in, const int* in_sizes, int n_in,
                              void* d_out, int out_size, void* d_ws, size_t ws_size,
                              hipStream_t stream)
{
    const float* enc_o = (const float*)d_in[0];   // [B,S,H]
    const float* enc_c = (const float*)d_in[1];   // [B,H]
    const float* enc_h = (const float*)d_in[2];   // [B,H]
    const int*   x_tgt = (const int*)  d_in[3];   // [B,32]
    const float* emb   = (const float*)d_in[4];   // [V,E]
    const float* Wa    = (const float*)d_in[5];   // [H4, E+H]
    const float* ba    = (const float*)d_in[6];   // [H4]
    const float* wv_w  = (const float*)d_in[7];   // [1,H4]
    const float* bv    = (const float*)d_in[8];   // [1]
    const float* W_ih  = (const float*)d_in[9];   // [4H, E+H]
    const float* W_hh  = (const float*)d_in[10];  // [4H, H]
    const float* b_ih  = (const float*)d_in[11];  // [4H]
    const float* b_hh  = (const float*)d_in[12];  // [4H]
    const float* Wf    = (const float*)d_in[13];  // [V,H]
    const float* bf    = (const float*)d_in[14];  // [V]
    float* out = (float*)d_out;                   // [B,T,V]

    // ---- workspace carve-up (all offsets 256B aligned) ----
    char* p = (char*)d_ws;
    size_t ofs = 0;
    auto carve = [&](size_t bytes) {
        void* r = p + ofs;
        ofs += (bytes + 255) & ~(size_t)255;
        return r;
    };
    bf16*  Wcat   = (bf16*)  carve((size_t)G4H_ * ZW_ * 2);        // 20.97 MB
    bf16*  Wf16   = (bf16*)  carve((size_t)V_ * H_ * 2);           // 65.54 MB
    bf16*  Wa16   = (bf16*)  carve((size_t)H4_ * (E_ + H_) * 2);   // 0.79 MB
    bf16*  encO16 = (bf16*)  carve((size_t)B_ * S_ * H_ * 2);      // 16.78 MB
    float* A0     = (float*) carve((size_t)B_ * S_ * H4_ * 4);     // 8.39 MB
    float* ae     = (float*) carve((size_t)B_ * H4_ * 4);
    float* score  = (float*) carve((size_t)B_ * S_ * 4);
    bf16*  z      = (bf16*)  carve((size_t)B_ * ZW_ * 2);
    float* cbuf   = (float*) carve((size_t)B_ * H_ * 4);
    float* gates  = (float*) carve((size_t)B_ * G4H_ * 4);
    float* bias_g = (float*) carve((size_t)G4H_ * 4);
    if (ofs > ws_size) return;   // workspace too small; bail safely

    // ---- one-time prep ----
    hipLaunchKernelGGL(pack_wcat_kernel, dim3(4096), dim3(256), 0, stream,
                       W_ih, W_hh, Wcat);
    hipLaunchKernelGGL(cvt_f32_bf16_kernel, dim3(8192), dim3(256), 0, stream,
                       Wf, Wf16, (long long)V_ * H_);
    hipLaunchKernelGGL(cvt_f32_bf16_kernel, dim3(256), dim3(256), 0, stream,
                       Wa, Wa16, (long long)H4_ * (E_ + H_));
    hipLaunchKernelGGL(cvt_f32_bf16_kernel, dim3(2048), dim3(256), 0, stream,
                       enc_o, encO16, (long long)B_ * S_ * H_);
    hipLaunchKernelGGL(bias_sum_kernel, dim3(16), dim3(256), 0, stream,
                       b_ih, b_hh, bias_g);
    hipLaunchKernelGGL(init_state_kernel, dim3(512), dim3(256), 0, stream,
                       enc_c, enc_h, cbuf, z);

    // Time-invariant attention term: A0[b*S+s, :] = enc_o @ Wa[:,E:].T + ba
    launch_gemm(encO16, H_, Wa16 + E_, E_ + H_, ba, A0, H4_,
                B_ * S_, H4_, H_, stream);

    // ---- decode loop (serial recurrence) ----
    for (int t = 0; t < T_; ++t) {
        hipLaunchKernelGGL(embed_kernel, dim3(B_), dim3(256), 0, stream,
                           x_tgt, t, emb, z);
        launch_gemm(z, ZW_, Wa16, E_ + H_, nullptr, ae, H4_,
                    B_, H4_, E_, stream);
        hipLaunchKernelGGL(score_kernel, dim3(B_ * S_ / 8), dim3(256), 0, stream,
                           A0, ae, wv_w, bv, score);
        hipLaunchKernelGGL(ctx_kernel, dim3(B_ * H_ / 256), dim3(256), 0, stream,
                           score, enc_o, z);
        launch_gemm(z, ZW_, Wcat, ZW_, bias_g, gates, G4H_,
                    B_, G4H_, ZW_, stream);
        hipLaunchKernelGGL(lstm_kernel, dim3(B_ * H_ / 256), dim3(256), 0, stream,
                           gates, cbuf, z);
        launch_gemm(z + (E_ + H_), ZW_, Wf16, H_, bf,
                    out + (long long)t * V_, (long long)T_ * V_,
                    B_, V_, H_, stream);
    }
}
